// MANet_63213328663166
// MI455X (gfx1250) — matrix operand order repaired and checked by reading, not compiled, not run on hardware
//
#include <hip/hip_runtime.h>
#include <hip/hip_bf16.h>
#include <stdint.h>

typedef __attribute__((ext_vector_type(16))) _Float16 v16h;
typedef __attribute__((ext_vector_type(8)))  float    v8f;

#define BATCH 64
#define DMODEL 128
#define NNODES 2048
#define HEADS 4
#define DK 32
#define INV_SCALE 0.17677669529663687f  // 1/sqrt(32)

#if __has_builtin(__builtin_amdgcn_tensor_load_to_lds) && \
    __has_builtin(__builtin_amdgcn_s_wait_tensorcnt)
#define HAVE_TDM 1
#else
#define HAVE_TDM 0
#endif

// ---------------------------------------------------------------------------
// WMMA helper: D = A(16x32 f16) * B(32x16 f16) + C(16x16 f32)
// ---------------------------------------------------------------------------
__device__ __forceinline__ v8f wmma_f32_f16(v16h a, v16h b, v8f c) {
  return __builtin_amdgcn_wmma_f32_16x16x32_f16(false, a, false, b, (short)0, c,
                                                false, false);
}

// A-fragment from row-major f16 (rows = M, K contiguous).
// Layout: lane<16 -> M=lane, K 0..15; lane>=16 -> M=lane-16, K 16..31.
// Per lane the 16 halfs are two contiguous 16B runs -> 2 x b128 loads.
__device__ __forceinline__ v16h load_a_rowmajor(const _Float16* __restrict__ p,
                                                int row_stride, int m0, int k0,
                                                int lane) {
  union { v16h v; uint4 q[2]; } f;
  const int m  = m0 + (lane & 15);
  const int hi = lane >> 4;
  const uint4* row = reinterpret_cast<const uint4*>(p + (size_t)m * row_stride + k0);
  f.q[0] = row[2 * hi];       // halfs k0 + hi*16     .. +7
  f.q[1] = row[2 * hi + 1];   // halfs k0 + hi*16 + 8 .. +7
  return f.v;
}

// ---------------------------------------------------------------------------
// TDM: async-load a 32x32 f16 tile (row stride 32 elements) into LDS.
// D# built per CDNA5 ISA 8.3/8.4; issued by one wave, EXEC ignored.
// ---------------------------------------------------------------------------
#if HAVE_TDM
__device__ __forceinline__ void tdm_load_tile32x32_f16(_Float16* lds_dst,
                                                       const _Float16* gsrc) {
  typedef __attribute__((ext_vector_type(4))) unsigned int tdm_v4u;
  typedef __attribute__((ext_vector_type(8))) int tdm_v8i;
  typedef __attribute__((ext_vector_type(4))) int tdm_v4i;
  const uint64_t ga   = (uint64_t)(uintptr_t)gsrc;
  const uint32_t ldsb = (uint32_t)(uintptr_t)lds_dst;  // LDS aperture low bits
  tdm_v4u g0;
  g0[0] = 1u;                                            // count=1 (valid user D#)
  g0[1] = ldsb;                                          // lds_addr (bytes)
  g0[2] = (uint32_t)ga;                                  // global_addr[31:0]
  g0[3] = (uint32_t)((ga >> 32) & 0x01FFFFFFu) | (2u << 30);  // [56:32] | type=2
  tdm_v8i g1;
  g1[0] = (int)(1u << 16);    // workgroup_mask=0, data_size=1 (2 bytes)
  g1[1] = (int)(32u << 16);   // tensor_dim0 = 32 (lo16 @ [31:16])
  g1[2] = (int)(32u << 16);   // tensor_dim1 = 32 (lo16 @ [31:16]); dim0 hi = 0
  g1[3] = (int)(32u << 16);   // tile_dim0 = 32;  tensor_dim1 hi = 0
  g1[4] = 32;                 // tile_dim1 = 32; tile_dim2 = 0
  g1[5] = 32;                 // tensor_dim0_stride = 32 elements
  g1[6] = 0;                  // stride hi / dim1 stride
  g1[7] = 0;
  tdm_v4i gz = {0, 0, 0, 0};  // 2D tensor: groups 2/3 unused
#if defined(__clang_major__) && (__clang_major__ >= 23)
  tdm_v8i gz8 = {0, 0, 0, 0, 0, 0, 0, 0};
  __builtin_amdgcn_tensor_load_to_lds(g0, g1, gz, gz, gz8, 0);
#else
  __builtin_amdgcn_tensor_load_to_lds(g0, g1, gz, gz, 0);
#endif
}
#endif

// ---------------------------------------------------------------------------
// 0) convert the three 128x128 weight matrices to f16 (row-major [o][c])
// ---------------------------------------------------------------------------
__global__ __launch_bounds__(256)
void cvt_weights_kernel(const float* __restrict__ qw, const float* __restrict__ vw,
                        const float* __restrict__ cw, _Float16* __restrict__ qwh,
                        _Float16* __restrict__ vwh, _Float16* __restrict__ cwh) {
  int i = blockIdx.x * 256 + threadIdx.x;
  if (i < DMODEL * DMODEL) {
    qwh[i] = (_Float16)qw[i];
    vwh[i] = (_Float16)vw[i];
    cwh[i] = (_Float16)cw[i];
  }
}

// ---------------------------------------------------------------------------
// 1) key_ = softmax(memory / sqrt(dk), axis=-1)  -> f16  [H][N][32]
// ---------------------------------------------------------------------------
__global__ __launch_bounds__(128)
void key_softmax_kernel(const float* __restrict__ memory, _Float16* __restrict__ ksm) {
  int row  = blockIdx.x * 4 + (threadIdx.x >> 5);   // h*N + n, 8192 rows
  int lane = threadIdx.x & 31;
  float v = memory[(size_t)row * DK + lane] * INV_SCALE;
  float mx = v;
#pragma unroll
  for (int off = 16; off > 0; off >>= 1) mx = fmaxf(mx, __shfl_xor(mx, off, 32));
  float ev = __expf(v - mx);
  float s = ev;
#pragma unroll
  for (int off = 16; off > 0; off >>= 1) s += __shfl_xor(s, off, 32);
  ksm[(size_t)row * DK + lane] = (_Float16)(ev / s);
}

// ---------------------------------------------------------------------------
// 2) bias_dyn[n,c] = softmax(relu(nv1 @ nv2), axis=-1)[n,:] @ bias_pool
// ---------------------------------------------------------------------------
__global__ __launch_bounds__(256)
void bias_dyn_kernel(const float* __restrict__ nv1, const float* __restrict__ nv2,
                     const float* __restrict__ bias_pool, float* __restrict__ bias_dyn) {
  __shared__ float e[NNODES];
  __shared__ float red[8];
  __shared__ float pd[8][32];
  __shared__ float sh_mx, sh_sum;
  const int n = blockIdx.x;
  const int t = threadIdx.x;

  float a[10];
#pragma unroll
  for (int k = 0; k < 10; ++k) a[k] = nv1[n * 10 + k];

  float mx = -1e30f;
  for (int m = t; m < NNODES; m += 256) {
    float s = 0.f;
#pragma unroll
    for (int k = 0; k < 10; ++k) s += a[k] * nv2[k * NNODES + m];
    s = fmaxf(s, 0.f);
    e[m] = s;
    mx = fmaxf(mx, s);
  }
#pragma unroll
  for (int off = 16; off > 0; off >>= 1) mx = fmaxf(mx, __shfl_xor(mx, off, 32));
  if ((t & 31) == 0) red[t >> 5] = mx;
  __syncthreads();
  if (t == 0) {
    float m2 = red[0];
#pragma unroll
    for (int i = 1; i < 8; ++i) m2 = fmaxf(m2, red[i]);
    sh_mx = m2;
  }
  __syncthreads();
  mx = sh_mx;
  float sum = 0.f;
  for (int m = t; m < NNODES; m += 256) {
    float ev = __expf(e[m] - mx);
    e[m] = ev;
    sum += ev;
  }
  __syncthreads();
#pragma unroll
  for (int off = 16; off > 0; off >>= 1) sum += __shfl_xor(sum, off, 32);
  if ((t & 31) == 0) red[t >> 5] = sum;
  __syncthreads();
  if (t == 0) {
    float s2 = 0.f;
#pragma unroll
    for (int i = 0; i < 8; ++i) s2 += red[i];
    sh_sum = s2;
  }
  __syncthreads();
  const float inv = 1.f / sh_sum;
  const int c = t & 31, g = t >> 5;
  float acc = 0.f;
  for (int m = g * 256; m < (g + 1) * 256; ++m) acc += e[m] * bias_pool[m * DK + c];
  pd[g][c] = acc;
  __syncthreads();
  if (t < 32) {
    float s3 = 0.f;
#pragma unroll
    for (int gg = 0; gg < 8; ++gg) s3 += pd[gg][c];
    bias_dyn[n * DK + c] = s3 * inv;
  }
}

// ---------------------------------------------------------------------------
// 3) QV kernel: q = softmax(relu(Wq x + bq)/scale, dk), v = relu(Wv x + bv)
// ---------------------------------------------------------------------------
__global__ __launch_bounds__(128)
void qv_kernel(const float* __restrict__ x, const _Float16* __restrict__ qwh,
               const float* __restrict__ qb, const _Float16* __restrict__ vwh,
               const float* __restrict__ vb, _Float16* __restrict__ qs,
               _Float16* __restrict__ vs) {
  const int lane = threadIdx.x & 31;
  const int wave = threadIdx.x >> 5;
  const int b    = blockIdx.x >> 5;                       // 64 batches
  const int n0   = ((blockIdx.x & 31) * 4 + wave) * 16;   // 2048/16 tiles per b
  const int lo   = lane & 15;
  const int hi   = lane >> 4;
  const int n    = n0 + lo;

  // B fragments of X: B is 32x16 (K=c, N=col); lane: N=lo, K = 16*hi + j
  union BF { v16h v; _Float16 h[16]; } bf[4];
#pragma unroll
  for (int kc = 0; kc < 4; ++kc) {
#pragma unroll
    for (int j = 0; j < 16; ++j) {
      int c = kc * 32 + hi * 16 + j;
      bf[kc].h[j] = (_Float16)x[(size_t)(b * DMODEL + c) * NNODES + n];
    }
  }

#pragma unroll
  for (int h = 0; h < HEADS; ++h) {
    const size_t rowbase = ((size_t)(b * HEADS + h) * NNODES + n) * DK;
    union { _Float16 hh[8]; uint4 u; } pk;

    // ---- Q: two 16-row tiles (m = 2h, 2h+1) over K = 128 ----
    v8f q0 = {}, q1 = {};
#pragma unroll
    for (int kc = 0; kc < 4; ++kc) {
      q0 = wmma_f32_f16(load_a_rowmajor(qwh, DMODEL, (2 * h) * 16, kc * 32, lane), bf[kc].v, q0);
      q1 = wmma_f32_f16(load_a_rowmajor(qwh, DMODEL, (2 * h + 1) * 16, kc * 32, lane), bf[kc].v, q1);
    }
    float qv[16];
    float mx = -1e30f;
#pragma unroll
    for (int r = 0; r < 8; ++r) {
      int o0 = (2 * h) * 16 + r + hi * 8;
      int o1 = (2 * h + 1) * 16 + r + hi * 8;
      float a0 = fmaxf(q0[r] + qb[o0], 0.f);
      float a1 = fmaxf(q1[r] + qb[o1], 0.f);
      qv[r] = a0; qv[8 + r] = a1;
      mx = fmaxf(mx, fmaxf(a0, a1));
    }
    mx = fmaxf(mx, __shfl_xor(mx, 16, 32));   // lanes l, l^16 share column n
    float sum = 0.f;
#pragma unroll
    for (int r = 0; r < 16; ++r) { qv[r] = __expf((qv[r] - mx) * INV_SCALE); sum += qv[r]; }
    sum += __shfl_xor(sum, 16, 32);
    const float inv = 1.f / sum;
#pragma unroll
    for (int r = 0; r < 8; ++r) pk.hh[r] = (_Float16)(qv[r] * inv);
    *reinterpret_cast<uint4*>(qs + rowbase + hi * 8) = pk.u;         // dk 0..15
#pragma unroll
    for (int r = 0; r < 8; ++r) pk.hh[r] = (_Float16)(qv[8 + r] * inv);
    *reinterpret_cast<uint4*>(qs + rowbase + 16 + hi * 8) = pk.u;    // dk 16..31

    // ---- V ----
    v8f v0 = {}, v1 = {};
#pragma unroll
    for (int kc = 0; kc < 4; ++kc) {
      v0 = wmma_f32_f16(load_a_rowmajor(vwh, DMODEL, (2 * h) * 16, kc * 32, lane), bf[kc].v, v0);
      v1 = wmma_f32_f16(load_a_rowmajor(vwh, DMODEL, (2 * h + 1) * 16, kc * 32, lane), bf[kc].v, v1);
    }
#pragma unroll
    for (int r = 0; r < 8; ++r)
      pk.hh[r] = (_Float16)fmaxf(v0[r] + vb[(2 * h) * 16 + r + hi * 8], 0.f);
    *reinterpret_cast<uint4*>(vs + rowbase + hi * 8) = pk.u;
#pragma unroll
    for (int r = 0; r < 8; ++r)
      pk.hh[r] = (_Float16)fmaxf(v1[r] + vb[(2 * h + 1) * 16 + r + hi * 8], 0.f);
    *reinterpret_cast<uint4*>(vs + rowbase + 16 + hi * 8) = pk.u;
  }
}

// ---------------------------------------------------------------------------
// 4) kv[b,h,x,y] = sum_n key_[h,n,x] * v[b,h,n,y]
//    32-row chunks staged in LDS (TDM async tensor load when available),
//    4 waves per block, each owns one 16x16 output tile.
// ---------------------------------------------------------------------------
__global__ __launch_bounds__(128)
void kv_kernel(const _Float16* __restrict__ ksm, const _Float16* __restrict__ vs,
               float* __restrict__ kvws) {
  __shared__ __align__(16) _Float16 kbuf[32 * 32];
  __shared__ __align__(16) _Float16 vbuf[32 * 32];
  const int bh   = blockIdx.x;           // b*4 + h
  const int h    = bh & 3;
  const int wave = threadIdx.x >> 5;
  const int lane = threadIdx.x & 31;
  const int mt = wave >> 1, yt = wave & 1;
  const int lo = lane & 15, hi = lane >> 4;

  const _Float16* kbase = ksm + (size_t)h * NNODES * DK;
  const _Float16* vbase = vs + (size_t)bh * NNODES * DK;
  const int xrow = mt * 16 + lo;
  const int y    = yt * 16 + lo;

  v8f acc = {};
  for (int nb = 0; nb < NNODES; nb += 32) {
#if HAVE_TDM
    if (wave == 0) {
      tdm_load_tile32x32_f16(kbuf, kbase + (size_t)nb * DK);
      tdm_load_tile32x32_f16(vbuf, vbase + (size_t)nb * DK);
      __builtin_amdgcn_s_wait_tensorcnt(0);
    }
#else
    {
      const int t = threadIdx.x;   // 128 threads x 16B == one 2KB tile each
      reinterpret_cast<uint4*>(kbuf)[t] =
          reinterpret_cast<const uint4*>(kbase + (size_t)nb * DK)[t];
      reinterpret_cast<uint4*>(vbuf)[t] =
          reinterpret_cast<const uint4*>(vbase + (size_t)nb * DK)[t];
    }
#endif
    __syncthreads();
    union { v16h v; _Float16 h[16]; } af, bf;
#pragma unroll
    for (int vv = 0; vv < 8; ++vv) {
      int k0 = 2 * vv + hi * 16;     // K index = n offset
      af.h[2 * vv]     = kbuf[(k0) * DK + xrow];
      af.h[2 * vv + 1] = kbuf[(k0 + 1) * DK + xrow];
    }
#pragma unroll
    for (int j = 0; j < 16; ++j) bf.h[j] = vbuf[(hi * 16 + j) * DK + y];
    acc = wmma_f32_f16(af.v, bf.v, acc);
    __syncthreads();
  }
  float* out = kvws + (size_t)bh * DK * DK;
#pragma unroll
  for (int r = 0; r < 8; ++r) out[(mt * 16 + r + hi * 8) * DK + y] = acc[r];
}

// ---------------------------------------------------------------------------
// 5) output: attn = Q@kv + S*v + bias_dyn ; transpose through LDS ;
//    out = relu(Wc @ attn + bc) * (aff_w + 1) + aff_b
// ---------------------------------------------------------------------------
__global__ __launch_bounds__(256)
void out_kernel(const _Float16* __restrict__ qs, const _Float16* __restrict__ vs,
                const float* __restrict__ kvws, const float* __restrict__ bias_dyn,
                const float* __restrict__ wpool, const _Float16* __restrict__ cwh,
                const float* __restrict__ cb, const float* __restrict__ aff_w,
                const float* __restrict__ aff_b, float* __restrict__ out) {
  __shared__ _Float16 kvh[HEADS * DK * DK];
  __shared__ __align__(16) _Float16 T[8][DMODEL * 16];   // per-wave [d][col]

  const int b    = blockIdx.x >> 4;
  const int tg   = blockIdx.x & 15;
  const int wave = threadIdx.x >> 5;
  const int lane = threadIdx.x & 31;
  const int lo = lane & 15, hi = lane >> 4;

  const float* kvb = kvws + (size_t)b * HEADS * DK * DK;
  for (int i = threadIdx.x; i < HEADS * DK * DK; i += 256) kvh[i] = (_Float16)kvb[i];
  __syncthreads();

  float S = 0.f;
#pragma unroll
  for (int i = 0; i < 9; ++i) S += wpool[i];   // sum(weights_pool); Aapt rows sum to 1

  const int n0 = (tg * 8 + wave) * 16;
  _Float16* Tw = T[wave];

#pragma unroll
  for (int h = 0; h < HEADS; ++h) {
    // A fragment from q_sm: rows = n, K = dk -> two b128 loads per lane
    v16h af = load_a_rowmajor(qs + (size_t)(b * HEADS + h) * NNODES * DK,
                              DK, n0, 0, lane);
#pragma unroll
    for (int yt = 0; yt < 2; ++yt) {
      union { v16h v; _Float16 h[16]; } bf;
      const int y = yt * 16 + lo;
#pragma unroll
      for (int j = 0; j < 16; ++j) bf.h[j] = kvh[(h * DK + hi * 16 + j) * DK + y];
      v8f acc = {};
      acc = wmma_f32_f16(af, bf.v, acc);
      // attn_dyn = S*value + bias_dyn, then stash transposed tile in LDS
      const int d = h * DK + y;                 // output channel after merge
      union { _Float16 hh[8]; uint4 u; } pk;
#pragma unroll
      for (int r = 0; r < 8; ++r) {
        const int n = n0 + r + hi * 8;
        float vv2 = (float)vs[((size_t)(b * HEADS + h) * NNODES + n) * DK + y];
        pk.hh[r] = (_Float16)(acc[r] + S * vv2 + bias_dyn[n * DK + y]);
      }
      *reinterpret_cast<uint4*>(&Tw[d * 16 + hi * 8]) = pk.u;
    }
  }
  __syncthreads();   // uniform; publish per-wave LDS tiles

  // final conv: B fragments gathered from LDS tile (K = d, N = col)
  union { v16h v; _Float16 h[16]; } bf3[4];
#pragma unroll
  for (int kc = 0; kc < 4; ++kc) {
#pragma unroll
    for (int j = 0; j < 16; ++j) bf3[kc].h[j] = Tw[(kc * 32 + hi * 16 + j) * 16 + lo];
  }
  const int n = n0 + lo;
#pragma unroll
  for (int m = 0; m < 8; ++m) {
    v8f acc = {};
#pragma unroll
    for (int kc = 0; kc < 4; ++kc)
      acc = wmma_f32_f16(load_a_rowmajor(cwh, DMODEL, m * 16, kc * 32, lane), bf3[kc].v, acc);
#pragma unroll
    for (int r = 0; r < 8; ++r) {
      const int o = m * 16 + r + hi * 8;
      float val = fmaxf(acc[r] + cb[o], 0.f);
      const size_t ai = (size_t)o * NNODES + n;
      out[((size_t)b * DMODEL + o) * NNODES + n] = val * (aff_w[ai] + 1.f) + aff_b[ai];
    }
  }
}

// ---------------------------------------------------------------------------
extern "C" void kernel_launch(void* const* d_in, const int* in_sizes, int n_in,
                              void* d_out, int out_size, void* d_ws, size_t ws_size,
                              hipStream_t stream) {
  const float* x      = (const float*)d_in[0];
  const float* q_w    = (const float*)d_in[1];
  const float* q_b    = (const float*)d_in[2];
  const float* v_w    = (const float*)d_in[3];
  const float* v_b    = (const float*)d_in[4];
  const float* c_w    = (const float*)d_in[5];
  const float* c_b    = (const float*)d_in[6];
  const float* memory = (const float*)d_in[7];
  const float* nv1    = (const float*)d_in[8];
  const float* nv2    = (const float*)d_in[9];
  const float* wpool  = (const float*)d_in[10];
  const float* bpool  = (const float*)d_in[11];
  const float* aff_w  = (const float*)d_in[12];
  const float* aff_b  = (const float*)d_in[13];
  float* outp = (float*)d_out;

  // workspace carve (256B aligned regions)
  char* p = (char*)d_ws;
  const size_t qs_bytes  = (size_t)BATCH * HEADS * NNODES * DK * 2;   // 33.5 MB
  const size_t ksm_bytes = (size_t)HEADS * NNODES * DK * 2;
  const size_t kv_bytes  = (size_t)BATCH * HEADS * DK * DK * 4;
  const size_t bd_bytes  = (size_t)NNODES * DK * 4;
  const size_t w_bytes   = (size_t)DMODEL * DMODEL * 2;

  _Float16* qs  = (_Float16*)p; p += qs_bytes;
  _Float16* vsw = (_Float16*)p; p += qs_bytes;
  _Float16* ksm = (_Float16*)p; p += ksm_bytes;
  float*    kvw = (float*)p;    p += kv_bytes;
  float*    bdn = (float*)p;    p += bd_bytes;
  _Float16* qwh = (_Float16*)p; p += w_bytes;
  _Float16* vwh = (_Float16*)p; p += w_bytes;
  _Float16* cwh = (_Float16*)p; p += w_bytes;

  cvt_weights_kernel<<<dim3(64), dim3(256), 0, stream>>>(q_w, v_w, c_w, qwh, vwh, cwh);
  key_softmax_kernel<<<dim3(HEADS * NNODES / 4), dim3(128), 0, stream>>>(memory, ksm);
  bias_dyn_kernel<<<dim3(NNODES), dim3(256), 0, stream>>>(nv1, nv2, bpool, bdn);
  qv_kernel<<<dim3(BATCH * 32), dim3(128), 0, stream>>>(x, qwh, q_b, vwh, v_b, qs, vsw);
  kv_kernel<<<dim3(BATCH * HEADS), dim3(128), 0, stream>>>(ksm, vsw, kvw);
  out_kernel<<<dim3(BATCH * 16), dim3(256), 0, stream>>>(qs, vsw, kvw, bdn, wpool,
                                                         cwh, c_b, aff_w, aff_b, outp);
}
